// GNNEncoder_26706106646645
// MI455X (gfx1250) — matrix operand order repaired and checked by reading, not compiled, not run on hardware
//
#include <hip/hip_runtime.h>
#include <hip/hip_bf16.h>

// ---------------------------------------------------------------------------
// GNN encoder for MI455X (gfx1250, wave32, WMMA).
//   Layer: agg_{r,i} = scatter_add(x[src]*c{r,i}, dst)
//          y = relu( [agg_r|agg_i|x] @ [wr;-wi;lw^T] + (cb+lb) )
// GEMM on v_wmma_f32_16x16x32_bf16 with 2-term bf16 split of fp32 operands
// (3 WMMAs per 16x16x32 step -> ~16 mantissa bits, fp32 accumulate).
// All tile shapes are template constants so accumulators stay in VGPRs and
// the ct loop fully unrolls (previous round spilled acc[] to scratch).
// ---------------------------------------------------------------------------

typedef __attribute__((ext_vector_type(16))) __bf16    v16bf;
typedef __attribute__((ext_vector_type(8)))  float     v8f;
typedef __attribute__((ext_vector_type(4)))  unsigned  v4u;
typedef __attribute__((ext_vector_type(8)))  unsigned  v8u;

#define NNODE_PAD 128   // padded row stride for ws buffers
#define TILE_M    128
#define LDS_KS    40    // ushort row stride: 80 B = 16B-aligned, bank-spread

__device__ __forceinline__ unsigned pack_hilo(float a) {
    // [15:0] = truncate-to-bf16(a), [31:16] = bf16(residual). ~16 mantissa bits.
    unsigned fb = __float_as_uint(a);
    unsigned hb = fb & 0xFFFF0000u;
    float lo = a - __uint_as_float(hb);      // exact
    return (hb >> 16) | (__float_as_uint(lo) & 0xFFFF0000u);
}

// Build a v16bf fragment from two contiguous 16-byte LDS runs (4 dwords each).
__device__ __forceinline__ v16bf frag16(const unsigned short* p0,
                                        const unsigned short* p1) {
    v4u a = *(const v4u*)p0;
    v4u b = *(const v4u*)p1;
    v8u u;
    u[0] = a[0]; u[1] = a[1]; u[2] = a[2]; u[3] = a[3];
    u[4] = b[0]; u[5] = b[1]; u[6] = b[2]; u[7] = b[3];
    return __builtin_bit_cast(v16bf, u);
}

// ---------------------------------------------------------------------------
__global__ void zero_f32(float* __restrict__ p, long long n) {
    long long i = (long long)blockIdx.x * blockDim.x + threadIdx.x;
    if (i < n) p[i] = 0.0f;
}

// Pack B_cat = [wr ; -wi ; lw^T] (Kpad x Cout) into hi/lo-bf16 u32 words,
// and fuse bias = cb + lb.
__global__ void prep_b(const float* __restrict__ wr, const float* __restrict__ wi,
                       const float* __restrict__ lw, const float* __restrict__ cb,
                       const float* __restrict__ lb,
                       unsigned* __restrict__ bpack, float* __restrict__ bias,
                       int Cin, int Cout, int Kpad) {
    int idx = blockIdx.x * blockDim.x + threadIdx.x;
    if (idx >= Kpad * Cout) return;
    int k = idx / Cout, col = idx - k * Cout;
    float v = 0.0f;
    if (k < Cin)            v =  wr[(size_t)k * Cout + col];
    else if (k < 2 * Cin)   v = -wi[(size_t)(k - Cin) * Cout + col];
    else if (k < 3 * Cin)   v =  lw[(size_t)col * Cin + (k - 2 * Cin)];
    bpack[idx] = pack_hilo(v);
    if (k == 0) bias[col] = cb[col] + lb[col];
}

// One wave32 per edge: gather x[src] row, scale by complex edge coeff,
// scatter-add into agg_r/agg_i rows (L2-resident f32 atomics).
__global__ __launch_bounds__(256)
void edge_scatter(const float* __restrict__ x, const int* __restrict__ ei,
                  const float* __restrict__ w, const float* __restrict__ sim,
                  float* __restrict__ agg_r, float* __restrict__ agg_i,
                  int Cin, int E) {
    int wave = threadIdx.x >> 5;
    int lane = threadIdx.x & 31;
    int e = blockIdx.x * 8 + wave;
    if (e >= E) return;
    int src = ei[e];
    int dst = ei[E + e];
    float we = w[e];
    float cr = we * sim[2 * e + 0];
    float ci = we * sim[2 * e + 1];
    const float* xs = x + (size_t)src * Cin;
    float* ar = agg_r + (size_t)dst * Cin;
    float* ai = agg_i + (size_t)dst * Cin;
    for (int c = lane; c < Cin; c += 32) {
        float xv = xs[c];
        __hip_atomic_fetch_add(&ar[c], xv * cr, __ATOMIC_RELAXED, __HIP_MEMORY_SCOPE_AGENT);
        __hip_atomic_fetch_add(&ai[c], xv * ci, __ATOMIC_RELAXED, __HIP_MEMORY_SCOPE_AGENT);
    }
}

template <int CIN>
__device__ __forceinline__ float load_a(const float* __restrict__ aggr,
                                        const float* __restrict__ aggi,
                                        const float* __restrict__ xin,
                                        int row, int kg, int N) {
    if (row >= N) return 0.0f;
    if (kg < CIN)          return aggr[(size_t)row * CIN + kg];
    if (kg < 2 * CIN)      return aggi[(size_t)row * CIN + (kg - CIN)];
    if (kg < 3 * CIN)      return xin [(size_t)row * CIN + (kg - 2 * CIN)];
    return 0.0f;
}

// Fused GEMM: out = relu([agg_r|agg_i|x] @ Bpack + bias).
// 256 threads = 8 waves; wave w owns rows [w*16, w*16+16) of a 128-row tile
// across all COUT columns. LDS holds separate hi/lo bf16 planes in fragment
// order so fragment build = two ds_load_b128 per operand half.
template <int CIN, int COUT>
__global__ __launch_bounds__(256)
void gemm_wmma(const float* __restrict__ aggr, const float* __restrict__ aggi,
               const float* __restrict__ xin, const unsigned* __restrict__ bpack,
               const float* __restrict__ bias, float* __restrict__ out, int N) {
    constexpr int KPAD = ((3 * CIN + 31) / 32) * 32;
    constexpr int NCT  = COUT / 16;

    __shared__ __attribute__((aligned(16))) unsigned short a_hi[TILE_M][LDS_KS];
    __shared__ __attribute__((aligned(16))) unsigned short a_lo[TILE_M][LDS_KS];
    __shared__ __attribute__((aligned(16))) unsigned short b_hi[COUT][LDS_KS];
    __shared__ __attribute__((aligned(16))) unsigned short b_lo[COUT][LDS_KS];

    const int tid  = threadIdx.x;
    const int wave = tid >> 5;
    const int lane = tid & 31;
    const int g    = lane >> 4;      // half-wave group
    const int m    = lane & 15;
    const int row0 = blockIdx.x * TILE_M;

    v8f acc[NCT];
    #pragma unroll
    for (int i = 0; i < NCT; ++i)
        acc[i] = (v8f){0.f, 0.f, 0.f, 0.f, 0.f, 0.f, 0.f, 0.f};

    for (int k0 = 0; k0 < KPAD; k0 += 32) {
        // --- stage A tile: 128 rows x 32 K, two K per thread-step ---
        #pragma unroll 1
        for (int i = tid; i < TILE_M * 16; i += 256) {
            int r = i >> 4, kp = (i & 15) * 2;
            unsigned p0 = pack_hilo(load_a<CIN>(aggr, aggi, xin, row0 + r, k0 + kp,     N));
            unsigned p1 = pack_hilo(load_a<CIN>(aggr, aggi, xin, row0 + r, k0 + kp + 1, N));
            *(unsigned*)&a_hi[r][kp] = (p0 & 0xFFFFu) | (p1 << 16);
            *(unsigned*)&a_lo[r][kp] = (p0 >> 16)     | (p1 & 0xFFFF0000u);
        }
        // --- stage B tile transposed: [col][K], two K per thread-step ---
        #pragma unroll 1
        for (int i = tid; i < 16 * COUT; i += 256) {
            int col = i % COUT, kp = (i / COUT) * 2;
            unsigned p0 = bpack[(size_t)(k0 + kp)     * COUT + col];
            unsigned p1 = bpack[(size_t)(k0 + kp + 1) * COUT + col];
            *(unsigned*)&b_hi[col][kp] = (p0 & 0xFFFFu) | (p1 << 16);
            *(unsigned*)&b_lo[col][kp] = (p0 >> 16)     | (p1 & 0xFFFF0000u);
        }
        __syncthreads();

        // --- A fragments (ISA 16-bit 16x32 A layout: VGPR v = K 8g+2v,+1 (v<4), K 8g+2v+8,+1 (v>=4)) ---
        const int ar = wave * 16 + m;
        v16bf ahi = frag16(&a_hi[ar][g * 8], &a_hi[ar][16 + g * 8]);
        v16bf alo = frag16(&a_lo[ar][g * 8], &a_lo[ar][16 + g * 8]);

        #pragma unroll
        for (int ct = 0; ct < NCT; ++ct) {
            const int col = ct * 16 + m;
            v16bf bhi = frag16(&b_hi[col][g * 8], &b_hi[col][16 + g * 8]);
            v16bf blo = frag16(&b_lo[col][g * 8], &b_lo[col][16 + g * 8]);
            acc[ct] = __builtin_amdgcn_wmma_f32_16x16x32_bf16(
                false, ahi, false, bhi, (short)0, acc[ct], false, false);
            acc[ct] = __builtin_amdgcn_wmma_f32_16x16x32_bf16(
                false, ahi, false, blo, (short)0, acc[ct], false, false);
            acc[ct] = __builtin_amdgcn_wmma_f32_16x16x32_bf16(
                false, alo, false, bhi, (short)0, acc[ct], false, false);
        }
        __syncthreads();
    }

    // --- bias + ReLU epilogue (ISA 32-bit 16x16 C/D layout) ---
    #pragma unroll
    for (int ct = 0; ct < NCT; ++ct) {
        int col = ct * 16 + m;
        float bs = bias[col];
        #pragma unroll
        for (int r = 0; r < 8; ++r) {
            int row = row0 + wave * 16 + (g << 3) + r;
            if (row < N) {
                float v = acc[ct][r] + bs;
                out[(size_t)row * COUT + col] = v > 0.f ? v : 0.f;
            }
        }
    }
}

// ---------------------------------------------------------------------------
static void run_layer(const float* xin, int Cin, int Cout,
                      const float* wr, const float* wi, const float* cb,
                      const float* lw, const float* lb,
                      const int* ei, const float* w, const float* sim,
                      float* agg_r, float* agg_i, unsigned* bpack, float* bias,
                      float* out, int N, int E, hipStream_t stream) {
    int Kpad = ((3 * Cin + 31) / 32) * 32;

    long long nz = 2LL * N * NNODE_PAD;  // agg_r and agg_i are contiguous
    zero_f32<<<(int)((nz + 255) / 256), 256, 0, stream>>>(agg_r, nz);

    int nb = Kpad * Cout;
    prep_b<<<(nb + 255) / 256, 256, 0, stream>>>(wr, wi, lw, cb, lb,
                                                 bpack, bias, Cin, Cout, Kpad);

    edge_scatter<<<(E + 7) / 8, 256, 0, stream>>>(xin, ei, w, sim,
                                                  agg_r, agg_i, Cin, E);

    int grid = (N + TILE_M - 1) / TILE_M;
    if (Cin == 75)
        gemm_wmma<75, 128><<<grid, 256, 0, stream>>>(agg_r, agg_i, xin, bpack, bias, out, N);
    else if (Cout == 128)
        gemm_wmma<128, 128><<<grid, 256, 0, stream>>>(agg_r, agg_i, xin, bpack, bias, out, N);
    else
        gemm_wmma<128, 32><<<grid, 256, 0, stream>>>(agg_r, agg_i, xin, bpack, bias, out, N);
}

extern "C" void kernel_launch(void* const* d_in, const int* in_sizes, int n_in,
                              void* d_out, int out_size, void* d_ws, size_t ws_size,
                              hipStream_t stream) {
    const float* x   = (const float*)d_in[0];
    const int*   ei  = (const int*)  d_in[1];
    const float* w   = (const float*)d_in[2];
    const float* sim = (const float*)d_in[3];

    const int IN_DIM = 75;
    const int N = in_sizes[0] / IN_DIM;
    const int E = in_sizes[2];

    // Workspace layout (floats): X1 | X2 | AGG_R | AGG_I | BPACK | BIAS
    float*    X1    = (float*)d_ws;
    float*    X2    = X1 + (size_t)N * NNODE_PAD;
    float*    AGG_R = X2 + (size_t)N * NNODE_PAD;
    float*    AGG_I = AGG_R + (size_t)N * NNODE_PAD;
    unsigned* BPACK = (unsigned*)(AGG_I + (size_t)N * NNODE_PAD);
    float*    BIAS  = (float*)(BPACK + 384 * 128);

    // layer 0: 75 -> 128
    run_layer(x, 75, 128,
              (const float*)d_in[4], (const float*)d_in[5], (const float*)d_in[6],
              (const float*)d_in[7], (const float*)d_in[8],
              ei, w, sim, AGG_R, AGG_I, BPACK, BIAS, X1, N, E, stream);
    // layer 1: 128 -> 128
    run_layer(X1, 128, 128,
              (const float*)d_in[9], (const float*)d_in[10], (const float*)d_in[11],
              (const float*)d_in[12], (const float*)d_in[13],
              ei, w, sim, AGG_R, AGG_I, BPACK, BIAS, X2, N, E, stream);
    // layer 2: 128 -> 32 (bottleneck, writes d_out)
    run_layer(X2, 128, 32,
              (const float*)d_in[14], (const float*)d_in[15], (const float*)d_in[16],
              (const float*)d_in[17], (const float*)d_in[18],
              ei, w, sim, AGG_R, AGG_I, BPACK, BIAS, (float*)d_out, N, E, stream);
}
